// MatchLayer_78847009620112
// MI455X (gfx1250) — matrix operand order repaired and checked by reading, not compiled, not run on hardware
//
#include <hip/hip_runtime.h>

typedef __attribute__((ext_vector_type(8)))  __bf16 v8bf;
typedef __attribute__((ext_vector_type(16))) __bf16 v16bf;
typedef __attribute__((ext_vector_type(8)))  float  v8f;

#if __has_builtin(__builtin_amdgcn_sched_group_barrier)
#define SGB(mask, n, id) __builtin_amdgcn_sched_group_barrier((mask), (n), (id))
#else
#define SGB(mask, n, id)
#endif

static __device__ __forceinline__ __bf16 f2bf(float f) {
  union { float f; unsigned u; } uf; uf.f = f;
  unsigned r = uf.u + 0x7FFFu + ((uf.u >> 16) & 1u);   // round-to-nearest-even
  union { unsigned short s; __bf16 b; } ub;
  ub.s = (unsigned short)(r >> 16);
  return ub.b;
}

// ---------------- f32 -> bf16 elementwise (8 elems / thread) ----------------
__global__ void convert_f32_bf16(const float* __restrict__ in,
                                 __bf16* __restrict__ out, long n) {
  long i = ((long)blockIdx.x * blockDim.x + threadIdx.x) * 8;
  if (i >= n) return;
  float4 f0 = *(const float4*)(in + i);
  float4 f1 = *(const float4*)(in + i + 4);
  v8bf o;
  o[0] = f2bf(f0.x); o[1] = f2bf(f0.y); o[2] = f2bf(f0.z); o[3] = f2bf(f0.w);
  o[4] = f2bf(f1.x); o[5] = f2bf(f1.y); o[6] = f2bf(f1.z); o[7] = f2bf(f1.w);
  *(v8bf*)(out + i) = o;
}

// ---------------- f32 [R][C] -> bf16 [C][R] tiled transpose ----------------
__global__ void transpose_convert(const float* __restrict__ in,
                                  __bf16* __restrict__ out, int R, int C) {
  __shared__ float tile[32][33];
  int c0 = blockIdx.x * 32, r0 = blockIdx.y * 32;
  int tx = threadIdx.x, ty = threadIdx.y;           // block (32,8)
  #pragma unroll
  for (int i = 0; i < 32; i += 8)
    tile[ty + i][tx] = in[(long)(r0 + ty + i) * C + c0 + tx];
  __syncthreads();
  #pragma unroll
  for (int i = 0; i < 32; i += 8)
    out[(long)(c0 + ty + i) * R + r0 + tx] = f2bf(tile[tx][ty + i]);
}

// ---------------- bf16 WMMA GEMM: D[m][n] = sum_k A[m][k] * Bs[n][k] -------
// A: [M][K] row-major bf16.  Bs: [N][K] row-major bf16 (i.e. B^T storage).
// Block = 8 waves -> 128(M) x 128(N).  B-panel (128 x 32, 8 KB) staged per
// K-step into double-buffered LDS via async global->LDS copies, in WMMA
// B-fragment order [tile t][chunk c][lane][16B] (conflict-free ds_load_b128).
// A rows prefetched one K-step ahead; sched_group_barrier enforces the
// software pipeline  VMEMx2, DSx2, {DSx2, WMMA}x7, WMMA  per K-step.
// OUT_MODE 0: f32  D[m][n] (ldd)   1: bf16 D[m][n] (ldd)   2: bf16 D[n][m] (ldd)
template<int OUT_MODE>
__global__ __launch_bounds__(256) void gemm_bf16_wmma(
    const __bf16* __restrict__ A, const __bf16* __restrict__ Bs,
    void* __restrict__ Dv, int M, int N, int K, int ldd,
    long sA, long sB, long sD) {
  __shared__ alignas(16) __bf16 Bsh[2][4096];   // 2 x 8 KB B-panel buffers

  const int tid  = threadIdx.x;
  const int lane = tid & 31;
  const int wave = tid >> 5;
  const int hf   = lane >> 4;       // half-wave select
  const int row  = lane & 15;
  const int m0 = (blockIdx.x * 8 + wave) * 16;
  const int n0 = blockIdx.y * 128;

  const __bf16* Ab = A  + (long)blockIdx.z * sA;
  const __bf16* Bb = Bs + (long)blockIdx.z * sB;

  // ---- staging setup: thread tid owns fragment (t = tid>>5, l = tid&31) ----
  const int st = tid >> 5, sl = tid & 31;
  // fragment (t,l) = Bs[n0 + t*16 + (l&15)][k0 + (l>>4)*8 .. +7, k0 + 16 + (l>>4)*8 .. +7]  (32 B)
  const __bf16* gsrc = Bb + (long)(n0 + st * 16 + (sl & 15)) * K + (sl >> 4) * 8;
  const unsigned ldsBase = (unsigned)(unsigned long long)&Bsh[0][0];
  const unsigned ldsA16 = ldsBase + (unsigned)(((st * 2 + 0) * 32 + sl) * 16);
  const unsigned ldsB16 = ldsBase + (unsigned)(((st * 2 + 1) * 32 + sl) * 16);

  auto stage = [&](int k0, int s) {
    unsigned l0 = ldsA16 + (unsigned)s * 8192u;
    unsigned l1 = ldsB16 + (unsigned)s * 8192u;
    unsigned long long g0 = (unsigned long long)(gsrc + k0);
    unsigned long long g1 = (unsigned long long)(gsrc + k0 + 16);
    asm volatile(
        "global_load_async_to_lds_b128 %0, %2, off\n\t"
        "global_load_async_to_lds_b128 %1, %3, off"
        :: "v"(l0), "v"(l1), "v"(g0), "v"(g1)
        : "memory");
  };

  v8f acc[8] = {};
  const __bf16* aRow = Ab + (long)(m0 + row) * K;

  stage(0, 0);
  // A fragment for k0=0 (16-bit A 16x32 layout: half 0 K {0..7,16..23}, half 1 {8..15,24..31})
  v8bf aCurLo = *(const v8bf*)(aRow + hf * 8);
  v8bf aCurHi = *(const v8bf*)(aRow + 16 + hf * 8);

  int s = 0;
  for (int k0 = 0; k0 < K; k0 += 32, s ^= 1) {
    // buffer s ready (our asyncs done) + whole block past last reads of s^1
    asm volatile("s_wait_asynccnt 0x0" ::: "memory");
    __syncthreads();

    // next-K offset, clamped on last iteration (dead but valid work -> one BB)
    const int kN = (k0 + 32 < K) ? (k0 + 32) : 0;
    stage(kN, s ^ 1);

    // prefetch next A fragment: consumed next iteration -> L2 latency hidden
    v8bf aNxtLo = *(const v8bf*)(aRow + kN + hf * 8);
    v8bf aNxtHi = *(const v8bf*)(aRow + kN + 16 + hf * 8);

    v16bf a = __builtin_shufflevector(aCurLo, aCurHi,
        0,1,2,3,4,5,6,7,8,9,10,11,12,13,14,15);

    // B fragments: two-deep rotation so wmma(t) overlaps ds loads of t+1
    v8bf blo = *(const v8bf*)(&Bsh[s][(0 * 32 + lane) * 8]);
    v8bf bhi = *(const v8bf*)(&Bsh[s][(1 * 32 + lane) * 8]);
    #pragma unroll
    for (int t = 0; t < 8; ++t) {
      v8bf nlo = blo, nhi = bhi;
      if (t < 7) {
        nlo = *(const v8bf*)(&Bsh[s][((t * 2 + 2) * 32 + lane) * 8]);
        nhi = *(const v8bf*)(&Bsh[s][((t * 2 + 3) * 32 + lane) * 8]);
      }
      v16bf b = __builtin_shufflevector(blo, bhi,
          0,1,2,3,4,5,6,7,8,9,10,11,12,13,14,15);
      acc[t] = __builtin_amdgcn_wmma_f32_16x16x32_bf16(
          false, a, false, b, (short)0, acc[t], false, false);
      blo = nlo; bhi = nhi;
    }
    aCurLo = aNxtLo; aCurHi = aNxtHi;

    // enforce pipelined schedule: A loads early, DS pair t+1 before WMMA t
    SGB(0x020, 2, 0);                 // 2 VMEM reads (A prefetch)
    SGB(0x100, 2, 0);                 // DS read x2   (tile 0)
    #pragma unroll
    for (int t = 0; t < 7; ++t) {
      SGB(0x100, 2, 0);               // DS read x2   (tile t+1)
      SGB(0x008, 1, 0);               // WMMA         (tile t)
    }
    SGB(0x008, 1, 0);                 // WMMA         (tile 7)
  }

  // C/D layout: VGPR r, lane -> row M = r + 8*hf, col N = row
  if (OUT_MODE == 0) {
    float* D = (float*)Dv + (long)blockIdx.z * sD;
    #pragma unroll
    for (int t = 0; t < 8; ++t)
      #pragma unroll
      for (int r = 0; r < 8; ++r)
        D[(long)(m0 + r + 8 * hf) * ldd + n0 + t * 16 + row] = acc[t][r];
  } else if (OUT_MODE == 1) {
    __bf16* D = (__bf16*)Dv + (long)blockIdx.z * sD;
    #pragma unroll
    for (int t = 0; t < 8; ++t)
      #pragma unroll
      for (int r = 0; r < 8; ++r)
        D[(long)(m0 + r + 8 * hf) * ldd + n0 + t * 16 + row] = f2bf(acc[t][r]);
  } else {
    __bf16* D = (__bf16*)Dv + (long)blockIdx.z * sD;
    #pragma unroll
    for (int t = 0; t < 8; ++t) {
      v8bf p;
      #pragma unroll
      for (int r = 0; r < 8; ++r) p[r] = f2bf(acc[t][r]);
      *(v8bf*)(D + (long)(n0 + t * 16 + row) * ldd + m0 + 8 * hf) = p;  // packed b128
    }
  }
}

// ---------------- row softmax: f32 [rows][1024] -> bf16 [rows][1024] -------
__global__ __launch_bounds__(256) void softmax_rows(
    const float* __restrict__ S, __bf16* __restrict__ P, int rows) {
  const int lane = threadIdx.x & 31;
  const int wave = threadIdx.x >> 5;
  const int r = blockIdx.x * 8 + wave;
  if (r >= rows) return;
  const float* srow = S + (long)r * 1024;
  __bf16* prow = P + (long)r * 1024;
  float v[32];
  float mx = -3.402823466e38f;
  #pragma unroll
  for (int i = 0; i < 32; ++i) { v[i] = srow[lane + i * 32]; mx = fmaxf(mx, v[i]); }
  #pragma unroll
  for (int o = 16; o > 0; o >>= 1) mx = fmaxf(mx, __shfl_xor(mx, o, 32));
  float sum = 0.f;
  #pragma unroll
  for (int i = 0; i < 32; ++i) { v[i] = __expf(v[i] - mx); sum += v[i]; }
  #pragma unroll
  for (int o = 16; o > 0; o >>= 1) sum += __shfl_xor(sum, o, 32);
  float inv = 1.0f / sum;
  #pragma unroll
  for (int i = 0; i < 32; ++i) prow[lane + i * 32] = f2bf(v[i] * inv);
}

// ---------------------------------------------------------------------------
extern "C" void kernel_launch(void* const* d_in, const int* in_sizes, int n_in,
                              void* d_out, int out_size, void* d_ws, size_t ws_size,
                              hipStream_t stream) {
  (void)in_sizes; (void)n_in; (void)out_size; (void)ws_size;
  const float* q  = (const float*)d_in[0];   // [8,1024,1024]
  const float* c  = (const float*)d_in[1];   // [8,2048,1024]
  const float* wq = (const float*)d_in[2];   // [1024,1024]
  const float* wk = (const float*)d_in[3];
  const float* wv = (const float*)d_in[4];
  float* out = (float*)d_out;                // [8,2048,1024] f32

  const long B = 8, LQ = 1024, LC = 2048, D = 1024, H = 1024;
  const size_t MB = 1u << 20;
  char* ws = (char*)d_ws;
  __bf16* wqT = (__bf16*)(ws + 0 * MB);      //  2 MB  [H][D]
  __bf16* wkT = (__bf16*)(ws + 2 * MB);      //  2 MB
  __bf16* wvT = (__bf16*)(ws + 4 * MB);      //  2 MB
  __bf16* cB  = (__bf16*)(ws + 6 * MB);      // 32 MB  [B*LC][D]  (reused as P)
  __bf16* qB  = (__bf16*)(ws + 38 * MB);     // 16 MB  [B*LQ][D]
  __bf16* Qb  = (__bf16*)(ws + 54 * MB);     // 32 MB  [B*LC][H]
  __bf16* Kb  = (__bf16*)(ws + 86 * MB);     // 16 MB  [B*LQ][H]
  __bf16* VT  = (__bf16*)(ws + 102 * MB);    // 16 MB  [B][H][LQ]
  float*  S   = (float*)(ws + 118 * MB);     // 64 MB  [B][LC][LQ]
  __bf16* P   = cB;                          // 32 MB  [B][LC][LQ]

  // 1) precision conversion + weight transposes
  convert_f32_bf16<<<(B * LC * D) / (8 * 256), 256, 0, stream>>>(c, cB, B * LC * D);
  convert_f32_bf16<<<(B * LQ * D) / (8 * 256), 256, 0, stream>>>(q, qB, B * LQ * D);
  dim3 tb(32, 8), tg(32, 32);
  transpose_convert<<<tg, tb, 0, stream>>>(wq, wqT, (int)D, (int)H);
  transpose_convert<<<tg, tb, 0, stream>>>(wk, wkT, (int)D, (int)H);
  transpose_convert<<<tg, tb, 0, stream>>>(wv, wvT, (int)D, (int)H);

  // 2) projections: Q = c@wq (bf16), K = q@wk (bf16), V^T = (q@wv)^T (bf16 packed)
  gemm_bf16_wmma<1><<<dim3(128, 8, 1), 256, 0, stream>>>(
      cB, wqT, Qb, (int)(B * LC), (int)H, (int)D, (int)H, 0, 0, 0);
  gemm_bf16_wmma<1><<<dim3(64, 8, 1), 256, 0, stream>>>(
      qB, wkT, Kb, (int)(B * LQ), (int)H, (int)D, (int)H, 0, 0, 0);
  gemm_bf16_wmma<2><<<dim3(8, 8, 8), 256, 0, stream>>>(
      qB, wvT, VT, (int)LQ, (int)H, (int)D, (int)LQ, LQ * D, 0, H * LQ);

  // 3) scores S[b] = Q[b] @ K[b]^T  (f32)
  gemm_bf16_wmma<0><<<dim3(16, 8, 8), 256, 0, stream>>>(
      Qb, Kb, S, (int)LC, (int)LQ, (int)H, (int)LQ, LC * H, LQ * H, LC * LQ);

  // 4) softmax over LQ -> P (bf16)
  softmax_rows<<<(int)(B * LC / 8), 256, 0, stream>>>(S, P, (int)(B * LC));

  // 5) out[b] = P[b] @ V[b]  (f32)
  gemm_bf16_wmma<0><<<dim3(16, 8, 8), 256, 0, stream>>>(
      P, VT, out, (int)LC, (int)H, (int)LQ, (int)H, LC * LQ, H * LQ, LC * H);
}
